// Net_24257975287857
// MI455X (gfx1250) — matrix operand order repaired and checked by reading, not compiled, not run on hardware
//
#include <hip/hip_runtime.h>
#include <hip/hip_bf16.h>

#define NN      100000
#define NE      1600000
#define CH      128
#define NCLS    41
#define NPAD    48
#define HCAT    384

typedef __attribute__((ext_vector_type(16))) __bf16 v16bf;
typedef __attribute__((ext_vector_type(8)))  __bf16 v8bf;
typedef __attribute__((ext_vector_type(8)))  float  v8f;

// ---- WMMA fragment loaders (contiguous-k layouts, wave32) ----
// A 16x32 bf16: lane holds row (lane&15); elems 0..15 -> K = kb+0..15,
// kb = (lane>>4)*16. p points at row + kb.
__device__ __forceinline__ v16bf frag_a(const __bf16* p) {
    v8bf lo = *(const v8bf*)p;
    v8bf hi = *(const v8bf*)(p + 8);
    v16bf r;
#pragma unroll
    for (int i = 0; i < 8; ++i) { r[i] = lo[i]; r[i + 8] = hi[i]; }
    return r;
}
// B 32x16 bf16: lane holds col (lane&15); elems 0..15 -> K = (lane>>4)*16 + e.
// p points at col-row + (lane>>4)*16 : one contiguous 32B load.
__device__ __forceinline__ v16bf frag_b(const __bf16* p) {
    return *(const v16bf*)p;
}

// ---- small helper kernels ----
__global__ void k_zero_f32(float* __restrict__ p, int n) {
    int i = blockIdx.x * blockDim.x + threadIdx.x;
    if (i < n) p[i] = 0.0f;
}

__global__ void k_f32_to_bf16(const float* __restrict__ in, __bf16* __restrict__ out, int n) {
    int i = blockIdx.x * blockDim.x + threadIdx.x;
    if (i < n) out[i] = (__bf16)in[i];
}

// Pad W_lin [41,384] -> [48,384] bf16, zero rows 41..47
__global__ void k_pack_wlin(const float* __restrict__ W, __bf16* __restrict__ O) {
    int i = blockIdx.x * blockDim.x + threadIdx.x;
    if (i >= NPAD * HCAT) return;
    int row = i / HCAT, col = i - row * HCAT;
    O[i] = (row < NCLS) ? (__bf16)W[row * HCAT + col] : (__bf16)0.0f;
}

// ---- edge scatter: one wave per edge, lane covers 4 channels ----
__global__ void k_scatter_f32(const float* __restrict__ x,
                              const int* __restrict__ src, const int* __restrict__ dst,
                              const float* __restrict__ ew, float* __restrict__ agg) {
    unsigned t = blockIdx.x * blockDim.x + threadIdx.x;
    unsigned e = t >> 5;
    if (e >= NE) return;
    int c4 = (t & 31) << 2;
    int s = src[e], d = dst[e];
    float w = ew[e];
    float4 v = *(const float4*)(x + (size_t)s * CH + c4);
    float* o = agg + (size_t)d * CH + c4;
    __hip_atomic_fetch_add(o + 0, v.x * w, __ATOMIC_RELAXED, __HIP_MEMORY_SCOPE_AGENT);
    __hip_atomic_fetch_add(o + 1, v.y * w, __ATOMIC_RELAXED, __HIP_MEMORY_SCOPE_AGENT);
    __hip_atomic_fetch_add(o + 2, v.z * w, __ATOMIC_RELAXED, __HIP_MEMORY_SCOPE_AGENT);
    __hip_atomic_fetch_add(o + 3, v.w * w, __ATOMIC_RELAXED, __HIP_MEMORY_SCOPE_AGENT);
}

__global__ void k_scatter_bf16(const __bf16* __restrict__ x,
                               const int* __restrict__ src, const int* __restrict__ dst,
                               const float* __restrict__ ew, float* __restrict__ agg) {
    unsigned t = blockIdx.x * blockDim.x + threadIdx.x;
    unsigned e = t >> 5;
    if (e >= NE) return;
    int c4 = (t & 31) << 2;
    int s = src[e], d = dst[e];
    float w = ew[e];
    typedef __attribute__((ext_vector_type(4))) __bf16 v4bf;
    v4bf v = *(const v4bf*)(x + (size_t)s * CH + c4);
    float* o = agg + (size_t)d * CH + c4;
    __hip_atomic_fetch_add(o + 0, (float)v[0] * w, __ATOMIC_RELAXED, __HIP_MEMORY_SCOPE_AGENT);
    __hip_atomic_fetch_add(o + 1, (float)v[1] * w, __ATOMIC_RELAXED, __HIP_MEMORY_SCOPE_AGENT);
    __hip_atomic_fetch_add(o + 2, (float)v[2] * w, __ATOMIC_RELAXED, __HIP_MEMORY_SCOPE_AGENT);
    __hip_atomic_fetch_add(o + 3, (float)v[3] * w, __ATOMIC_RELAXED, __HIP_MEMORY_SCOPE_AGENT);
}

// ---- fused layer GEMM: out = relu(agg @ Wrel^T + b + x @ Wroot^T), bf16 out
// grid.x = NN/16 (row tiles), block = 256 (8 waves -> 8 col tiles of 16)
__global__ void k_gemm_layer(const __bf16* __restrict__ Abf,   // agg  [NN,128]
                             const __bf16* __restrict__ Xbf,   // x_in [NN,128]
                             const __bf16* __restrict__ Wrel,  // [128,128] row-major [out][in]
                             const __bf16* __restrict__ Wroot, // [128,128]
                             const float*  __restrict__ bias,  // [128]
                             __bf16* __restrict__ Out) {       // [NN,128]
    int wave = threadIdx.x >> 5;
    int lane = threadIdx.x & 31;
    int mrow = blockIdx.x * 16;
    int ncol = wave * 16;
    int r    = lane & 15;
    int half = lane >> 4;

    const __bf16* arow = Abf  + (size_t)(mrow + r) * CH + half * 16;
    const __bf16* xrow = Xbf  + (size_t)(mrow + r) * CH + half * 16;
    const __bf16* wrl  = Wrel  + (size_t)(ncol + r) * CH + half * 16;
    const __bf16* wrt  = Wroot + (size_t)(ncol + r) * CH + half * 16;

    v8f c = {};
#pragma unroll
    for (int k0 = 0; k0 < CH; k0 += 32) {
        v16bf a = frag_a(arow + k0);
        v16bf b = frag_b(wrl + k0);
        c = __builtin_amdgcn_wmma_f32_16x16x32_bf16(false, a, false, b, (short)0, c, false, false);
    }
#pragma unroll
    for (int k0 = 0; k0 < CH; k0 += 32) {
        v16bf a = frag_a(xrow + k0);
        v16bf b = frag_b(wrt + k0);
        c = __builtin_amdgcn_wmma_f32_16x16x32_bf16(false, a, false, b, (short)0, c, false, false);
    }

    int n = ncol + r;
    float bn = bias[n];
#pragma unroll
    for (int v = 0; v < 8; ++v) {
        int m = mrow + v + half * 8;
        float val = c[v] + bn;
        val = val > 0.0f ? val : 0.0f;
        Out[(size_t)m * CH + n] = (__bf16)val;
    }
}

// ---- classifier GEMM: logits[NN,48] = [x1|x2|x3] @ Wlin^T + b
// grid.x = NN/16, block = 96 (3 waves -> 3 col tiles of 16, covers 48 cols)
__global__ void k_gemm_final(const __bf16* __restrict__ x1, const __bf16* __restrict__ x2,
                             const __bf16* __restrict__ x3,
                             const __bf16* __restrict__ Wlin, // [48,384] bf16, padded
                             const float* __restrict__ blin,  // [41]
                             float* __restrict__ logits) {    // [NN,48]
    int wave = threadIdx.x >> 5;
    int lane = threadIdx.x & 31;
    int mrow = blockIdx.x * 16;
    int ncol = wave * 16;
    int r    = lane & 15;
    int half = lane >> 4;

    const __bf16* xs[3] = { x1, x2, x3 };
    const __bf16* wrow = Wlin + (size_t)(ncol + r) * HCAT + half * 16;

    v8f c = {};
    for (int s = 0; s < 3; ++s) {
        const __bf16* arow = xs[s] + (size_t)(mrow + r) * CH + half * 16;
#pragma unroll
        for (int k0 = 0; k0 < CH; k0 += 32) {
            v16bf a = frag_a(arow + k0);
            v16bf b = frag_b(wrow + s * CH + k0);
            c = __builtin_amdgcn_wmma_f32_16x16x32_bf16(false, a, false, b, (short)0, c, false, false);
        }
    }

    int n = ncol + r;
    float bn = (n < NCLS) ? blin[n] : 0.0f;
#pragma unroll
    for (int v = 0; v < 8; ++v) {
        int m = mrow + v + half * 8;
        logits[(size_t)m * NPAD + n] = c[v] + bn;
    }
}

// ---- per-node log-softmax over 41 classes ----
__global__ void k_logsoftmax(const float* __restrict__ logits, float* __restrict__ out) {
    int i = blockIdx.x * blockDim.x + threadIdx.x;
    if (i >= NN) return;
    const float* row = logits + (size_t)i * NPAD;
    float mx = row[0];
#pragma unroll 8
    for (int c = 1; c < NCLS; ++c) mx = fmaxf(mx, row[c]);
    float sum = 0.0f;
#pragma unroll 8
    for (int c = 0; c < NCLS; ++c) sum += expf(row[c] - mx);
    float ls = logf(sum);
    float* o = out + (size_t)i * NCLS;
#pragma unroll 8
    for (int c = 0; c < NCLS; ++c) o[c] = row[c] - mx - ls;
}

extern "C" void kernel_launch(void* const* d_in, const int* in_sizes, int n_in,
                              void* d_out, int out_size, void* d_ws, size_t ws_size,
                              hipStream_t stream) {
    const float* x0   = (const float*)d_in[0];
    const int*   ei   = (const int*)d_in[1];
    const float* ew   = (const float*)d_in[2];
    const float* Wrel[3]  = { (const float*)d_in[3], (const float*)d_in[6], (const float*)d_in[9]  };
    const float* brel[3]  = { (const float*)d_in[4], (const float*)d_in[7], (const float*)d_in[10] };
    const float* Wroot[3] = { (const float*)d_in[5], (const float*)d_in[8], (const float*)d_in[11] };
    const float* Wlin = (const float*)d_in[12];
    const float* blin = (const float*)d_in[13];
    const int* src = ei;
    const int* dst = ei + NE;

    // -------- workspace carve --------
    size_t off = 0;
    auto carve = [&](size_t bytes) -> void* {
        void* p = (char*)d_ws + off;
        off += (bytes + 255) & ~(size_t)255;
        return p;
    };
    float*  agg    = (float*)carve((size_t)NN * CH * 4);
    __bf16* agg_bf = (__bf16*)carve((size_t)NN * CH * 2);
    __bf16* x0_bf  = (__bf16*)carve((size_t)NN * CH * 2);
    __bf16* xl_bf[3];
    for (int l = 0; l < 3; ++l) xl_bf[l] = (__bf16*)carve((size_t)NN * CH * 2);
    float*  logits = (float*)carve((size_t)NN * NPAD * 4);
    __bf16* wrel_bf[3], *wroot_bf[3];
    for (int l = 0; l < 3; ++l) {
        wrel_bf[l]  = (__bf16*)carve((size_t)CH * CH * 2);
        wroot_bf[l] = (__bf16*)carve((size_t)CH * CH * 2);
    }
    __bf16* wlin_bf = (__bf16*)carve((size_t)NPAD * HCAT * 2);

    const int NCH = NN * CH;                 // 12.8M
    const int TPB = 256;

    // -------- weight + input conversion --------
    for (int l = 0; l < 3; ++l) {
        k_f32_to_bf16<<<(CH * CH + TPB - 1) / TPB, TPB, 0, stream>>>(Wrel[l],  wrel_bf[l],  CH * CH);
        k_f32_to_bf16<<<(CH * CH + TPB - 1) / TPB, TPB, 0, stream>>>(Wroot[l], wroot_bf[l], CH * CH);
    }
    k_pack_wlin<<<(NPAD * HCAT + TPB - 1) / TPB, TPB, 0, stream>>>(Wlin, wlin_bf);
    k_f32_to_bf16<<<(NCH + TPB - 1) / TPB, TPB, 0, stream>>>(x0, x0_bf, NCH);

    // -------- 3 GraphConv layers --------
    const unsigned scatter_blocks = ((unsigned)NE * 32u + TPB - 1) / TPB;  // 200000
    const __bf16* xin_bf = x0_bf;
    for (int l = 0; l < 3; ++l) {
        k_zero_f32<<<(NCH + TPB - 1) / TPB, TPB, 0, stream>>>(agg, NCH);
        if (l == 0)
            k_scatter_f32<<<scatter_blocks, TPB, 0, stream>>>(x0, src, dst, ew, agg);
        else
            k_scatter_bf16<<<scatter_blocks, TPB, 0, stream>>>(xin_bf, src, dst, ew, agg);
        k_f32_to_bf16<<<(NCH + TPB - 1) / TPB, TPB, 0, stream>>>(agg, agg_bf, NCH);
        k_gemm_layer<<<NN / 16, 256, 0, stream>>>(agg_bf, xin_bf, wrel_bf[l], wroot_bf[l],
                                                  brel[l], xl_bf[l]);
        xin_bf = xl_bf[l];
    }

    // -------- classifier + log-softmax --------
    k_gemm_final<<<NN / 16, 96, 0, stream>>>(xl_bf[0], xl_bf[1], xl_bf[2], wlin_bf, blin, logits);
    k_logsoftmax<<<(NN + TPB - 1) / TPB, TPB, 0, stream>>>(logits, (float*)d_out);
}
